// GUnet_84086869721206
// MI455X (gfx1250) — matrix operand order repaired and checked by reading, not compiled, not run on hardware
//
#include <hip/hip_runtime.h>
#include <hip/hip_bf16.h>
#include <math.h>

// ---------------------------------------------------------------------------
// Graph U-Net (dense adjacency) for MI455X / gfx1250, wave32 + WMMA bf16.
// Heavy ops (A@A augment, GCN aggregation) use v_wmma_f32_16x16x32_bf16.
// A-tile staging uses the gfx1250 async global->LDS path (ASYNCcnt).
// ---------------------------------------------------------------------------

typedef __attribute__((ext_vector_type(16))) __bf16 bf16x16;
typedef __attribute__((ext_vector_type(8)))  float  floatx8;

union FragB16 { bf16x16 v; uint4 u[2]; };

#define NN   4096
#define EE   131072
#define NFEAT 14
#define FDIM 16

// ============================ WMMA GEMM: C = A(bf16 MxK) @ B(bf16 KxN), f32 out
// 128x128 tile per 256-thread block (8 waves), K-step 32.
__global__ __launch_bounds__(256) void gemm_bf16_128(const __bf16* __restrict__ A,
                                                     const __bf16* __restrict__ B,
                                                     float* __restrict__ C,
                                                     int M, int N, int K) {
  __shared__ __bf16 Ash[128 * 32];   // [row][k]   row-major
  __shared__ __bf16 Bsh[128 * 32];   // [col][k]   K-transposed
  const int tid  = threadIdx.x;
  const int lane = tid & 31;
  const int wave = tid >> 5;
  const int bm = blockIdx.x * 128;
  const int bn = blockIdx.y * 128;

  floatx8 acc[8];
#pragma unroll
  for (int i = 0; i < 8; ++i) acc[i] = (floatx8){0.f,0.f,0.f,0.f,0.f,0.f,0.f,0.f};

  // per-thread staging coordinates (constant across K loop)
  const int ar  = tid >> 1;              // A row within strip
  const int acc_off = (tid & 1) * 16;    // A column chunk (16 elems = 32B)
  const unsigned a_lds = (unsigned)(uintptr_t)(Ash + ar * 32 + acc_off);

  for (int kt = 0; kt < K; kt += 32) {
    // ---- stage A strip 128x32 via async global->LDS (2 x b128 per lane).
    // INST_OFFSET applies to both the LDS and global addresses (ISA 10.x).
    {
      const __bf16* g0 = A + (size_t)(bm + ar) * K + kt + acc_off;
      asm volatile("global_load_async_to_lds_b128 %0, %1, off\n\t"
                   "global_load_async_to_lds_b128 %0, %1, off offset:16"
                   :: "v"(a_lds), "v"(g0) : "memory");
      if (kt + 32 < K)  // hint next strip into cache (global_prefetch_b8)
        __builtin_prefetch((const void*)(g0 + 32), 0, 1);
    }
    // ---- stage B strip 32x128, transposed into [col][k]
    {
      const int r  = tid >> 3;          // K row within strip
      const int cc = (tid & 7) * 16;    // column chunk
      const __bf16* src = B + (size_t)(kt + r) * N + bn + cc;
#pragma unroll
      for (int i = 0; i < 16; ++i) Bsh[(cc + i) * 32 + r] = src[i];
      if (kt + 32 < K)
        __builtin_prefetch((const void*)(B + (size_t)(kt + 32 + r) * N + bn + cc), 0, 1);
    }
    // async copies must land before the workgroup barrier releases readers
    asm volatile("s_wait_asynccnt 0" ::: "memory");
    __syncthreads();

    // ---- build A fragment (contiguous k = 16*(lane/16) + s, same order as B)
    const int row = wave * 16 + (lane & 15);
    const int klo = lane & 16;                 // 0 or 16
    FragB16 fa;
    fa.u[0] = *(const uint4*)(Ash + row * 32 + klo);
    fa.u[1] = *(const uint4*)(Ash + row * 32 + 8 + klo);

    const int coln = lane & 15;
    const int kb   = lane & 16;                // 0 or 16
#pragma unroll
    for (int nt = 0; nt < 8; ++nt) {
      FragB16 fb;
      const uint4* bp = (const uint4*)(Bsh + (nt * 16 + coln) * 32 + kb);
      fb.u[0] = bp[0];
      fb.u[1] = bp[1];
      acc[nt] = __builtin_amdgcn_wmma_f32_16x16x32_bf16(
          false, fa.v, false, fb.v, (short)0, acc[nt], false, false);
    }
    __syncthreads();
  }

  // ---- store C (ISA 32-bit C/D 16x16 lane layout)
  const int coln = lane & 15;
  const int mh   = (lane & 16) >> 1;           // 0 or 8
#pragma unroll
  for (int nt = 0; nt < 8; ++nt)
#pragma unroll
    for (int r = 0; r < 8; ++r)
      C[(size_t)(bm + wave * 16 + mh + r) * N + bn + nt * 16 + coln] = acc[nt][r];
}

// ============================ WMMA GCN aggregation: Z(kx16 f32) = Abf(kxk) @ Ybf(kx16)
__global__ __launch_bounds__(256) void gcn_agg_wmma(const __bf16* __restrict__ Abf,
                                                    const __bf16* __restrict__ Ybf,
                                                    float* __restrict__ Z, int Kdim) {
  __shared__ __bf16 Ysh[16 * 32];   // [col 16][k 32] transposed
  const int tid  = threadIdx.x;
  const int lane = tid & 31;
  const int wave = tid >> 5;
  const int bm = blockIdx.x * 128;

  floatx8 acc = (floatx8){0.f,0.f,0.f,0.f,0.f,0.f,0.f,0.f};

  for (int kt = 0; kt < Kdim; kt += 32) {
    {
      const int e0 = tid * 2;
      const int r0 = e0 >> 4, c0 = e0 & 15;
      Ysh[c0 * 32 + r0] = Ybf[(size_t)(kt + r0) * 16 + c0];
      const int e1 = e0 + 1;
      const int r1 = e1 >> 4, c1 = e1 & 15;
      Ysh[c1 * 32 + r1] = Ybf[(size_t)(kt + r1) * 16 + c1];
    }
    __syncthreads();

    const int row = bm + wave * 16 + (lane & 15);
    const int klo = lane & 16;
    FragB16 fa;
    fa.u[0] = *(const uint4*)(Abf + (size_t)row * Kdim + kt + klo);
    fa.u[1] = *(const uint4*)(Abf + (size_t)row * Kdim + kt + 8 + klo);
    FragB16 fb;
    const uint4* bp = (const uint4*)(Ysh + (lane & 15) * 32 + (lane & 16));
    fb.u[0] = bp[0];
    fb.u[1] = bp[1];
    acc = __builtin_amdgcn_wmma_f32_16x16x32_bf16(
        false, fa.v, false, fb.v, (short)0, acc, false, false);
    __syncthreads();
  }

  const int coln = lane & 15;
  const int mh   = (lane & 16) >> 1;
#pragma unroll
  for (int r = 0; r < 8; ++r)
    Z[(size_t)(bm + wave * 16 + mh + r) * 16 + coln] = acc[r];
}

// ============================ glue kernels
__global__ void zero_f32(float* p, size_t n) {
  for (size_t i = (size_t)blockIdx.x * blockDim.x + threadIdx.x; i < n;
       i += (size_t)gridDim.x * blockDim.x) p[i] = 0.f;
}

__global__ void build_edges(const int* __restrict__ ei, float* __restrict__ A, int e_count, int n) {
  int e = blockIdx.x * blockDim.x + threadIdx.x;
  if (e < e_count) {
    int s = ei[e];
    int d = ei[e_count + e];
    atomicAdd(&A[(size_t)s * n + d], 1.0f);
  }
}

// dis[i] = rsqrt(rowsum(A + selfloop-fix)) (0 if degree 0)
__global__ void deg_dis(const float* __restrict__ A, float* __restrict__ dis, int k) {
  int i = blockIdx.x * blockDim.x + threadIdx.x;
  if (i >= k) return;
  const float* row = A + (size_t)i * k;
  float s = 0.f;
  for (int j = 0; j < k; ++j) s += row[j];
  if (row[i] == 0.f) s += 1.f;
  dis[i] = (s > 0.f) ? rsqrtf(s) : 0.f;
}

// Abf = bf16(A + diag(where diag==0 -> 1))
__global__ void abf_hat(const float* __restrict__ A, __bf16* __restrict__ Abf, int k) {
  size_t idx = (size_t)blockIdx.x * blockDim.x + threadIdx.x;
  size_t n = (size_t)k * k;
  if (idx >= n) return;
  int i = (int)(idx / k), j = (int)(idx % k);
  float v = A[idx];
  if (i == j && v == 0.f) v = 1.f;
  Abf[idx] = (__bf16)v;
}

// Abf = bf16(A*(1-I) + I)   (augment pre-matrix)
__global__ void abf_aug(const float* __restrict__ A, __bf16* __restrict__ Abf, int k) {
  size_t idx = (size_t)blockIdx.x * blockDim.x + threadIdx.x;
  size_t n = (size_t)k * k;
  if (idx >= n) return;
  int i = (int)(idx / k), j = (int)(idx % k);
  Abf[idx] = (__bf16)((i == j) ? 1.0f : A[idx]);
}

__global__ void zero_diag(float* A, int k) {
  int i = blockIdx.x * blockDim.x + threadIdx.x;
  if (i < k) A[(size_t)i * k + i] = 0.f;
}

// Y(rows x 16) = X(rows x indim) @ W(indim x 16)
__global__ void xw_kernel(const float* __restrict__ X, const float* __restrict__ W,
                          float* __restrict__ Y, int rows, int indim) {
  int idx = blockIdx.x * blockDim.x + threadIdx.x;
  if (idx >= rows * FDIM) return;
  int r = idx >> 4, d = idx & 15;
  float s = 0.f;
  for (int i = 0; i < indim; ++i) s += X[(size_t)r * indim + i] * W[i * FDIM + d];
  Y[idx] = s;
}

__global__ void scale_bf(const float* __restrict__ Y, const float* __restrict__ dis,
                         __bf16* __restrict__ Ybf, int rows) {
  int idx = blockIdx.x * blockDim.x + threadIdx.x;
  if (idx >= rows * FDIM) return;
  int r = idx >> 4;
  Ybf[idx] = (__bf16)(Y[idx] * dis[r]);
}

__global__ void finalize_k(const float* __restrict__ Zin, const float* __restrict__ dis,
                           const float* __restrict__ b, float* __restrict__ out,
                           int rows, int dorelu) {
  int idx = blockIdx.x * blockDim.x + threadIdx.x;
  if (idx >= rows * FDIM) return;
  int r = idx >> 4, d = idx & 15;
  float v = Zin[idx] * dis[r] + b[d];
  if (dorelu && v < 0.f) v = 0.f;
  out[idx] = v;
}

__global__ void pnorm_k(const float* __restrict__ p, float* __restrict__ out) {
  if (threadIdx.x == 0 && blockIdx.x == 0) {
    float s = 0.f;
    for (int i = 0; i < FDIM; ++i) s += p[i] * p[i];
    out[0] = sqrtf(s);
  }
}

__global__ void score_k(const float* __restrict__ X, const float* __restrict__ p,
                        const float* __restrict__ pn, float* __restrict__ s, int rows) {
  int i = blockIdx.x * blockDim.x + threadIdx.x;
  if (i >= rows) return;
  float acc = 0.f;
  for (int d = 0; d < FDIM; ++d) acc += X[(size_t)i * FDIM + d] * p[d];
  s[i] = tanhf(acc / pn[0]);
}

// Full bitonic sort (descending, ties -> lower index first), single block.
#define SORTN 4096
__global__ __launch_bounds__(1024) void topk_sort(const float* __restrict__ score, int k,
                                                  int keep, int* __restrict__ perm,
                                                  float* __restrict__ vals) {
  __shared__ float key[SORTN];
  __shared__ int   idx[SORTN];
  const int tid = threadIdx.x;
  for (int i = tid; i < SORTN; i += 1024) {
    key[i] = (i < k) ? score[i] : -__builtin_inff();
    idx[i] = i;
  }
  __syncthreads();
  for (int sz = 2; sz <= SORTN; sz <<= 1) {
    for (int st = sz >> 1; st > 0; st >>= 1) {
      for (int i = tid; i < SORTN; i += 1024) {
        int l = i ^ st;
        if (l > i) {
          // gt(l,i): element l ranks before element i in descending order
          bool gt_li = (key[l] > key[i]) || (key[l] == key[i] && idx[l] < idx[i]);
          bool desc  = ((i & sz) == 0);
          if (desc ? gt_li : !gt_li) {
            float tk = key[i]; key[i] = key[l]; key[l] = tk;
            int   ti = idx[i]; idx[i] = idx[l]; idx[l] = ti;
          }
        }
      }
      __syncthreads();
    }
  }
  for (int i = tid; i < keep; i += 1024) {
    perm[i] = idx[i];
    vals[i] = key[i];
  }
}

__global__ void gather_A(const float* __restrict__ Aaug, const int* __restrict__ perm,
                         float* __restrict__ An, int kold, int knew) {
  size_t idx = (size_t)blockIdx.x * blockDim.x + threadIdx.x;
  size_t n = (size_t)knew * knew;
  if (idx >= n) return;
  int i = (int)(idx / knew), j = (int)(idx % knew);
  An[idx] = Aaug[(size_t)perm[i] * kold + perm[j]];
}

__global__ void gather_x(const float* __restrict__ X, const int* __restrict__ perm,
                         const float* __restrict__ vals, float* __restrict__ Xn, int knew) {
  int idx = blockIdx.x * blockDim.x + threadIdx.x;
  if (idx >= knew * FDIM) return;
  int i = idx >> 4, d = idx & 15;
  Xn[idx] = X[(size_t)perm[i] * FDIM + d] * vals[i];
}

__global__ void copy_f32(float* __restrict__ dst, const float* __restrict__ src, size_t n) {
  size_t i = (size_t)blockIdx.x * blockDim.x + threadIdx.x;
  if (i < n) dst[i] = src[i];
}

// unpool: X[perm[j]][:] += Xs[j][:]   (perm entries unique -> plain add)
__global__ void scatter_add(float* __restrict__ X, const int* __restrict__ perm,
                            const float* __restrict__ Xs, int ksmall) {
  int idx = blockIdx.x * blockDim.x + threadIdx.x;
  if (idx >= ksmall * FDIM) return;
  int j = idx >> 4, d = idx & 15;
  X[(size_t)perm[j] * FDIM + d] += Xs[idx];
}

// ---------------------------------------------------------------------------
extern "C" void kernel_launch(void* const* d_in, const int* in_sizes, int n_in,
                              void* d_out, int out_size, void* d_ws, size_t ws_size,
                              hipStream_t stream) {
  const float* x_in  = (const float*)d_in[0];    // 4096 x 14
  const int*   ei    = (const int*)d_in[1];      // 2 x E
  const float* W_d0  = (const float*)d_in[2];
  const float* b_d0  = (const float*)d_in[3];
  const float* W_d1  = (const float*)d_in[4];
  const float* b_d1  = (const float*)d_in[5];
  const float* W_d2  = (const float*)d_in[6];
  const float* b_d2  = (const float*)d_in[7];
  const float* W_d3  = (const float*)d_in[8];
  const float* b_d3  = (const float*)d_in[9];
  const float* p0    = (const float*)d_in[10];
  const float* p1    = (const float*)d_in[11];
  const float* p2    = (const float*)d_in[12];
  const float* W_u0  = (const float*)d_in[13];
  const float* b_u0  = (const float*)d_in[14];
  const float* W_u1  = (const float*)d_in[15];
  const float* b_u1  = (const float*)d_in[16];
  const float* W_u2  = (const float*)d_in[17];
  const float* b_u2  = (const float*)d_in[18];

  // ---- workspace carve-out
  char* w = (char*)d_ws;
  auto carve = [&](size_t bytes) -> char* {
    char* p = w;
    w += (bytes + 255) & ~(size_t)255;
    return p;
  };
  float*  A0   = (float*)carve((size_t)NN * NN * 4);          // level-0 adjacency (As[0])
  float*  Aaug = (float*)carve((size_t)NN * NN * 4);          // augment result (reused)
  __bf16* Abf  = (__bf16*)carve((size_t)NN * NN * 2);         // bf16 operand (reused)
  float*  A1   = (float*)carve((size_t)2048 * 2048 * 4);      // As[1]
  float*  A2b  = (float*)carve((size_t)1024 * 1024 * 4);      // As[2]
  float*  A3   = (float*)carve((size_t)512 * 512 * 4);
  float*  xs0  = (float*)carve((size_t)NN * FDIM * 4);
  float*  xs1  = (float*)carve((size_t)2048 * FDIM * 4);
  float*  xs2  = (float*)carve((size_t)1024 * FDIM * 4);
  float*  xA   = (float*)carve((size_t)NN * FDIM * 4);
  float*  xB   = (float*)carve((size_t)NN * FDIM * 4);
  float*  ybuf = (float*)carve((size_t)NN * FDIM * 4);
  float*  zbuf = (float*)carve((size_t)NN * FDIM * 4);
  __bf16* ybf  = (__bf16*)carve((size_t)NN * FDIM * 2);
  float*  dis  = (float*)carve((size_t)NN * 4);
  float*  scor = (float*)carve((size_t)NN * 4);
  float*  vals = (float*)carve((size_t)2048 * 4);
  int*    prm0 = (int*)carve((size_t)2048 * 4);
  int*    prm1 = (int*)carve((size_t)1024 * 4);
  int*    prm2 = (int*)carve((size_t)512 * 4);
  float*  pnv  = (float*)carve(256);

  auto g1 = [](size_t n) { return dim3((unsigned)((n + 255) / 256)); };

  // GCN: out = relu?( dis ⊙ (Âhat @ (dis ⊙ (xin@W))) + b )
  auto gcn = [&](int k, const float* A, const float* xin, int indim, const float* W,
                 const float* b, int relu, float* xout) {
    size_t kk = (size_t)k * k;
    abf_hat<<<g1(kk), 256, 0, stream>>>(A, Abf, k);
    deg_dis<<<g1(k), 256, 0, stream>>>(A, dis, k);
    xw_kernel<<<g1((size_t)k * FDIM), 256, 0, stream>>>(xin, W, ybuf, k, indim);
    scale_bf<<<g1((size_t)k * FDIM), 256, 0, stream>>>(ybuf, dis, ybf, k);
    gcn_agg_wmma<<<dim3(k / 128), 256, 0, stream>>>(Abf, ybf, zbuf, k);
    finalize_k<<<g1((size_t)k * FDIM), 256, 0, stream>>>(zbuf, dis, b, xout, k, relu);
  };

  // augment: Aaug = ((A*(1-I)+I) @ (A*(1-I)+I)) * (1-I)
  auto augment = [&](int k, const float* A) {
    size_t kk = (size_t)k * k;
    abf_aug<<<g1(kk), 256, 0, stream>>>(A, Abf, k);
    gemm_bf16_128<<<dim3(k / 128, k / 128), 256, 0, stream>>>(Abf, Abf, Aaug, k, k, k);
    zero_diag<<<g1(k), 256, 0, stream>>>(Aaug, k);
  };

  auto pool = [&](int kold, int knew, const float* p, const float* xin, float* Anew,
                  int* perm, float* xout) {
    pnorm_k<<<1, 1, 0, stream>>>(p, pnv);
    score_k<<<g1(kold), 256, 0, stream>>>(xin, p, pnv, scor, kold);
    topk_sort<<<1, 1024, 0, stream>>>(scor, kold, knew, perm, vals);
    gather_A<<<g1((size_t)knew * knew), 256, 0, stream>>>(Aaug, perm, Anew, kold, knew);
    gather_x<<<g1((size_t)knew * FDIM), 256, 0, stream>>>(xin, perm, vals, xout, knew);
  };

  // ---- build dense adjacency
  zero_f32<<<dim3(4096), 256, 0, stream>>>(A0, (size_t)NN * NN);
  build_edges<<<g1(EE), 256, 0, stream>>>(ei, A0, EE, NN);

  // ---- first GCN (14 -> 16), relu; xs0 = x
  gcn(NN, A0, x_in, NFEAT, W_d0, b_d0, 1, xs0);

  // ---- down path
  augment(4096, A0);
  pool(4096, 2048, p0, xs0, A1, prm0, xA);
  gcn(2048, A1, xA, FDIM, W_d1, b_d1, 1, xs1);

  augment(2048, A1);
  pool(2048, 1024, p1, xs1, A2b, prm1, xA);
  gcn(1024, A2b, xA, FDIM, W_d2, b_d2, 1, xs2);

  augment(1024, A2b);
  pool(1024, 512, p2, xs2, A3, prm2, xA);
  gcn(512, A3, xA, FDIM, W_d3, b_d3, 1, xB);        // xB: 512x16

  // ---- up path
  copy_f32<<<g1((size_t)1024 * FDIM), 256, 0, stream>>>(xA, xs2, (size_t)1024 * FDIM);
  scatter_add<<<g1((size_t)512 * FDIM), 256, 0, stream>>>(xA, prm2, xB, 512);
  gcn(1024, A2b, xA, FDIM, W_u0, b_u0, 1, xB);      // xB: 1024x16

  copy_f32<<<g1((size_t)2048 * FDIM), 256, 0, stream>>>(xA, xs1, (size_t)2048 * FDIM);
  scatter_add<<<g1((size_t)1024 * FDIM), 256, 0, stream>>>(xA, prm1, xB, 1024);
  gcn(2048, A1, xA, FDIM, W_u1, b_u1, 1, xB);       // xB: 2048x16

  copy_f32<<<g1((size_t)4096 * FDIM), 256, 0, stream>>>(xA, xs0, (size_t)4096 * FDIM);
  scatter_add<<<g1((size_t)2048 * FDIM), 256, 0, stream>>>(xA, prm0, xB, 2048);
  gcn(4096, A0, xA, FDIM, W_u2, b_u2, 0, (float*)d_out);  // no relu on final layer

  (void)in_sizes; (void)n_in; (void)out_size; (void)ws_size;
}
